// FactorizedTransition_5935644803188
// MI455X (gfx1250) — matrix operand order repaired and checked by reading, compiled, not run on hardware
//
#include <hip/hip_runtime.h>
#include <stdint.h>

// ---------------------------------------------------------------------------
// Problem constants (from reference)
// ---------------------------------------------------------------------------
#define NUM_VARS 16
#define N_STATES 65536
#define S_STATES 2048
#define EMB 128
#define HID 128

typedef __attribute__((ext_vector_type(16))) _Float16     v16h;
typedef __attribute__((ext_vector_type(8)))  float        v8f;
typedef __attribute__((ext_vector_type(4)))  unsigned int v4u;
typedef __attribute__((ext_vector_type(4)))  float        v4f;

union FragH { v16h h; v4u u[2]; };

// ---------------------------------------------------------------------------
// WMMA fragment loaders (16-bit, 16x16x32, per CDNA5 ISA 7.12.2 layouts)
// A (16x32, MxK): lane<16 -> M=lane, K={kk+0..7, kk+16..23}
//                 lane>=16 -> M=lane-16, K={kk+8..15, kk+24..31}
// B (32x16, KxN): lane L -> N=L&15, K = {kk+(L>>4)*8 .. +7, kk+16+(L>>4)*8 .. +7} (same as A)
// ---------------------------------------------------------------------------
__device__ __forceinline__ void load_a_frags(FragH A[4], const _Float16* q, int s0, int lane) {
  int half = lane >> 4;
  int r    = lane & 15;
  const _Float16* base = q + (size_t)(s0 + r) * EMB + half * 8;
#pragma unroll
  for (int t = 0; t < 4; ++t) {
    A[t].u[0] = *(const v4u*)(base + t * 32);
    A[t].u[1] = *(const v4u*)(base + t * 32 + 16);
  }
}

__device__ __forceinline__ void load_b_frags(FragH B[4], const _Float16* kall, int n0, int lane) {
  int n  = n0 + (lane & 15);
  int kb = (lane >> 4) * 8;
  const _Float16* base = kall + (size_t)n * EMB + kb;
#pragma unroll
  for (int t = 0; t < 4; ++t) {
    B[t].u[0] = *(const v4u*)(base + t * 32);
    B[t].u[1] = *(const v4u*)(base + t * 32 + 16);
  }
}

__device__ __forceinline__ v8f wmma16(const FragH& a, const FragH& b, v8f c) {
  return __builtin_amdgcn_wmma_f32_16x16x32_f16(
      /*neg_a=*/false, a.h, /*neg_b=*/false, b.h,
      /*c_mod=*/(short)0, c, /*reuse_a=*/false, /*reuse_b=*/false);
}

// Online softmax element update with a SINGLE exp (one factor is always 1):
__device__ __forceinline__ void softmax_push(float& m, float& z, float l) {
  float d = m - l;                       // >0: keep scale; <0: rescale by exp(d)
  float t = __expf(-fabsf(d));
  z = (d < 0.0f) ? __fmaf_rn(z, t, 1.0f) : (z + t);
  m = fmaxf(m, l);
}

__device__ __forceinline__ void softmax_merge(float& m, float& z, float mo, float zo) {
  float mn = fmaxf(m, mo);
  z = z * __expf(m - mn) + zo * __expf(mo - mn);
  m = mn;
}

// ---------------------------------------------------------------------------
// K1: bit vectors -> indices (MSB-first)
// ---------------------------------------------------------------------------
__global__ void k_idcs(const int* __restrict__ bits, int* __restrict__ idcs) {
  int s = blockIdx.x * blockDim.x + threadIdx.x;
  if (s >= S_STATES) return;
  int v = 0;
#pragma unroll
  for (int b = 0; b < NUM_VARS; ++b) v = (v << 1) | (bits[s * NUM_VARS + b] & 1);
  idcs[s] = v;
}

// ---------------------------------------------------------------------------
// K2: q_sel[s,h] = state_emb[idcs[s]] . query_w[h] + query_b[h]   (-> f16)
// ---------------------------------------------------------------------------
__global__ __launch_bounds__(128) void k_qsel(const float* __restrict__ emb,
                                              const int* __restrict__ idcs,
                                              const float* __restrict__ qw,
                                              const float* __restrict__ qb,
                                              _Float16* __restrict__ q) {
  __shared__ float row[EMB];
  int s = blockIdx.x, h = threadIdx.x;
  int idx = idcs[s];
  row[h] = emb[(size_t)idx * EMB + h];
  __syncthreads();
  float acc = qb[h];
  const v4f* e = (const v4f*)row;
  const v4f* w = (const v4f*)(qw + (size_t)h * EMB);
#pragma unroll 8
  for (int k4 = 0; k4 < EMB / 4; ++k4) {
    v4f ev = e[k4], wv = w[k4];
    acc += ev[0] * wv[0] + ev[1] * wv[1] + ev[2] * wv[2] + ev[3] * wv[3];
  }
  q[(size_t)s * EMB + h] = (_Float16)acc;
}

// ---------------------------------------------------------------------------
// K3: k_all[n,h] = state_emb[n] . key_w[h] + key_b[h]   (-> f16)
// ---------------------------------------------------------------------------
__global__ __launch_bounds__(256) void k_kall(const float* __restrict__ emb,
                                              const float* __restrict__ kw,
                                              const float* __restrict__ kb,
                                              _Float16* __restrict__ kall) {
  __shared__ float E[32 * EMB];  // 16 KB
  int n0 = blockIdx.x * 32;
  for (int i = threadIdx.x; i < 32 * EMB; i += 256) E[i] = emb[(size_t)n0 * EMB + i];
  __syncthreads();
  for (int o = threadIdx.x; o < 32 * HID; o += 256) {
    int r = o >> 7, h = o & 127;
    float acc = kb[h];
    const v4f* e = (const v4f*)&E[r * EMB];
    const v4f* w = (const v4f*)(kw + (size_t)h * EMB);
#pragma unroll 8
    for (int k4 = 0; k4 < EMB / 4; ++k4) {
      v4f ev = e[k4], wv = w[k4];
      acc += ev[0] * wv[0] + ev[1] * wv[1] + ev[2] * wv[2] + ev[3] * wv[3];
    }
    kall[(size_t)(n0 + r) * EMB + h] = (_Float16)acc;
  }
}

// ---------------------------------------------------------------------------
// K4: row stats (online softmax).  Block = 16 rows of q, 8 waves split N.
// A fragments register-resident for the whole kernel; B fragments double-
// buffered (ping-pong, manual unroll-by-2) so loads for tile i+1 are in
// flight while WMMAs consume tile i.
// ---------------------------------------------------------------------------
__global__ __launch_bounds__(256) void k_rowstats(const _Float16* __restrict__ q,
                                                  const _Float16* __restrict__ kall,
                                                  float* __restrict__ m_out,
                                                  float* __restrict__ z_out) {
  int s0   = blockIdx.x * 16;
  int wave = threadIdx.x >> 5;
  int lane = threadIdx.x & 31;

  FragH A[4];
  load_a_frags(A, q, s0, lane);

  float m[8], z[8];
#pragma unroll
  for (int v = 0; v < 8; ++v) { m[v] = -3.0e38f; z[v] = 0.0f; }

  const int base = wave * 16;
  const int step = 128;                 // 8 waves * 16 cols
  FragH Ba[4], Bb[4];
  load_b_frags(Ba, kall, base, lane);

  for (int n = base; n < N_STATES; n += 2 * step) {
    // issue loads for tile i+1, then compute tile i
    load_b_frags(Bb, kall, n + step, lane);
    __builtin_prefetch(kall + (size_t)(n + 2 * step) * EMB, 0, 3);
    v8f c0 = {};
#pragma unroll
    for (int t = 0; t < 4; ++t) c0 = wmma16(A[t], Ba[t], c0);
#pragma unroll
    for (int v = 0; v < 8; ++v) softmax_push(m[v], z[v], c0[v]);

    // issue loads for tile i+2, then compute tile i+1
    if (n + 2 * step < N_STATES) load_b_frags(Ba, kall, n + 2 * step, lane);
    v8f c1 = {};
#pragma unroll
    for (int t = 0; t < 4; ++t) c1 = wmma16(A[t], Bb[t], c1);
#pragma unroll
    for (int v = 0; v < 8; ++v) softmax_push(m[v], z[v], c1[v]);
  }

  // reduce across the 16 column-lanes of each half (wave32 shuffles)
#pragma unroll
  for (int off = 1; off <= 8; off <<= 1) {
#pragma unroll
    for (int v = 0; v < 8; ++v) {
      float mo = __shfl_xor(m[v], off, 32);
      float zo = __shfl_xor(z[v], off, 32);
      softmax_merge(m[v], z[v], mo, zo);
    }
  }

  // lane 0 holds rows s0+0..7, lane 16 holds rows s0+8..15; merge 8 waves via LDS
  __shared__ float sm[8][16];
  __shared__ float sz[8][16];
  if ((lane & 15) == 0) {
#pragma unroll
    for (int v = 0; v < 8; ++v) {
      int r = v + (lane >> 4) * 8;
      sm[wave][r] = m[v];
      sz[wave][r] = z[v];
    }
  }
  __syncthreads();
  if (threadIdx.x < 16) {
    int r = threadIdx.x;
    float M = sm[0][r], Z = sz[0][r];
#pragma unroll
    for (int w = 1; w < 8; ++w) softmax_merge(M, Z, sm[w][r], sz[w][r]);
    m_out[s0 + r] = M;
    z_out[s0 + r] = Z;
  }
}

// ---------------------------------------------------------------------------
// K5: prior[n] = sum_s exp(l_{s,n} - sb[s]),  sb[s] = m_s + ln Z_s - ln belief_s
// Block owns 128 columns (wave -> 16 cols, B frags register-resident for the
// entire s loop).  A fragments double-buffered across s-tiles.  sb staged in
// LDS once, read as ds_load_b128 (v4f) pairs.
// ---------------------------------------------------------------------------
__global__ __launch_bounds__(256) void k_prior(const _Float16* __restrict__ q,
                                               const _Float16* __restrict__ kall,
                                               const float* __restrict__ belief,
                                               const float* __restrict__ m_in,
                                               const float* __restrict__ z_in,
                                               float* __restrict__ prior) {
  __shared__ float sb[S_STATES];        // 8 KB
  int wave = threadIdx.x >> 5;
  int lane = threadIdx.x & 31;
  int n0   = blockIdx.x * 128 + wave * 16;

  for (int i = threadIdx.x; i < S_STATES; i += 256)
    sb[i] = m_in[i] + __logf(z_in[i]) - __logf(belief[i]);
  __syncthreads();

  FragH B[4];                           // loop-invariant across the whole s loop
  load_b_frags(B, kall, n0, lane);

  FragH Aa[4], Ab[4];
  load_a_frags(Aa, q, 0, lane);

  float acc = 0.0f;
  const int rh = (lane >> 4) * 8;
  for (int s0 = 0; s0 < S_STATES; s0 += 32) {
    // loads for tile i+1 in flight while computing tile i
    load_a_frags(Ab, q, s0 + 16, lane);
    {
      v8f c = {};
#pragma unroll
      for (int t = 0; t < 4; ++t) c = wmma16(Aa[t], B[t], c);
      const v4f* p = (const v4f*)&sb[s0 + rh];
      v4f b0 = p[0], b1 = p[1];
#pragma unroll
      for (int v = 0; v < 4; ++v) acc += __expf(c[v] - b0[v]);
#pragma unroll
      for (int v = 0; v < 4; ++v) acc += __expf(c[v + 4] - b1[v]);
    }
    if (s0 + 32 < S_STATES) load_a_frags(Aa, q, s0 + 32, lane);
    {
      v8f c = {};
#pragma unroll
      for (int t = 0; t < 4; ++t) c = wmma16(Ab[t], B[t], c);
      const v4f* p = (const v4f*)&sb[s0 + 16 + rh];
      v4f b0 = p[0], b1 = p[1];
#pragma unroll
      for (int v = 0; v < 4; ++v) acc += __expf(c[v] - b0[v]);
#pragma unroll
      for (int v = 0; v < 4; ++v) acc += __expf(c[v + 4] - b1[v]);
    }
  }
  acc += __shfl_xor(acc, 16, 32);       // merge the two half-tiles (same column)
  if (lane < 16) prior[n0 + lane] = acc;
}

// ---------------------------------------------------------------------------
// K6: pack sort keys: (value_bits << 32) | ~index  (values >= 0)
// ---------------------------------------------------------------------------
__global__ void k_keys(const float* __restrict__ prior, unsigned long long* __restrict__ keys) {
  int i = blockIdx.x * blockDim.x + threadIdx.x;
  if (i >= N_STATES) return;
  unsigned vb = __float_as_uint(prior[i]);
  keys[i] = ((unsigned long long)vb << 32) | (unsigned)(~(unsigned)i);
}

// ---------------------------------------------------------------------------
// K7: one bitonic compare-exchange pass (descending)
// ---------------------------------------------------------------------------
__global__ void k_bitonic(unsigned long long* __restrict__ keys, int j, int k) {
  int i = blockIdx.x * blockDim.x + threadIdx.x;
  int ixj = i ^ j;
  if (ixj > i && i < N_STATES) {
    unsigned long long a = keys[i], b = keys[ixj];
    bool up = ((i & k) == 0);
    if (up ? (a < b) : (a > b)) { keys[i] = b; keys[ixj] = a; }
  }
}

// ---------------------------------------------------------------------------
// K8: sparsemax over the top-2048 (sorted desc) + decode indices to bits
// ---------------------------------------------------------------------------
__global__ __launch_bounds__(1024) void k_sparsemax(const unsigned long long* __restrict__ keys,
                                                    float* __restrict__ out,
                                                    int* __restrict__ bits_out) {
  __shared__ float a[S_STATES];
  __shared__ int kcnt;
  int t = threadIdx.x;
  int i0 = t, i1 = t + 1024;

  unsigned long long k0 = keys[i0], k1 = keys[i1];
  float v0 = __uint_as_float((unsigned)(k0 >> 32));
  float v1 = __uint_as_float((unsigned)(k1 >> 32));
  unsigned idx0 = ~(unsigned)(k0 & 0xffffffffu);
  unsigned idx1 = ~(unsigned)(k1 & 0xffffffffu);
#pragma unroll
  for (int b = 0; b < NUM_VARS; ++b) {
    bits_out[i0 * NUM_VARS + b] = (int)((idx0 >> (NUM_VARS - 1 - b)) & 1u);
    bits_out[i1 * NUM_VARS + b] = (int)((idx1 >> (NUM_VARS - 1 - b)) & 1u);
  }

  a[i0] = v0;
  a[i1] = v1;
  if (t == 0) kcnt = 0;
  __syncthreads();

  for (int off = 1; off < S_STATES; off <<= 1) {
    float t0 = (i0 >= off) ? a[i0 - off] : 0.0f;
    float t1 = (i1 >= off) ? a[i1 - off] : 0.0f;
    __syncthreads();
    a[i0] += t0;
    a[i1] += t1;
    __syncthreads();
  }

  int c = (int)((1.0f + (float)(i0 + 1) * v0) > a[i0]) +
          (int)((1.0f + (float)(i1 + 1) * v1) > a[i1]);
  atomicAdd(&kcnt, c);
  __syncthreads();

  int ks = kcnt;                        // support size >= 1 always
  float tau = (a[ks - 1] - 1.0f) / (float)ks;
  out[i0] = fmaxf(v0 - tau, 0.0f);
  out[i1] = fmaxf(v1 - tau, 0.0f);
}

// ---------------------------------------------------------------------------
// Host launcher
// ---------------------------------------------------------------------------
extern "C" void kernel_launch(void* const* d_in, const int* in_sizes, int n_in,
                              void* d_out, int out_size, void* d_ws, size_t ws_size,
                              hipStream_t stream) {
  const float* belief = (const float*)d_in[0];   // [1,2048]
  const int*   bits   = (const int*)d_in[1];     // [2048,16]
  const float* emb    = (const float*)d_in[2];   // [65536,128]
  const float* kw     = (const float*)d_in[3];   // [128,128]
  const float* kb     = (const float*)d_in[4];   // [128]
  const float* qw     = (const float*)d_in[5];   // [128,128]
  const float* qb     = (const float*)d_in[6];   // [128]

  uint8_t* ws = (uint8_t*)d_ws;
  _Float16* q_f16    = (_Float16*)(ws + 0);                 //   0.5 MB
  _Float16* kall_f16 = (_Float16*)(ws + 524288);            //  16.0 MB
  float*    m_buf    = (float*)(ws + 17301504);             //   8 KB
  float*    z_buf    = (float*)(ws + 17309696);             //   8 KB
  float*    prior    = (float*)(ws + 17317888);             // 256 KB
  unsigned long long* keys = (unsigned long long*)(ws + 17580032);  // 512 KB
  int*      idcs     = (int*)(ws + 18104320);               //   8 KB
  (void)ws_size; (void)in_sizes; (void)n_in; (void)out_size;

  float* out_vals = (float*)d_out;
  int*   out_bits = (int*)((float*)d_out + S_STATES);

  k_idcs<<<S_STATES / 256, 256, 0, stream>>>(bits, idcs);
  k_qsel<<<S_STATES, 128, 0, stream>>>(emb, idcs, qw, qb, q_f16);
  k_kall<<<N_STATES / 32, 256, 0, stream>>>(emb, kw, kb, kall_f16);

  k_rowstats<<<S_STATES / 16, 256, 0, stream>>>(q_f16, kall_f16, m_buf, z_buf);
  k_prior<<<N_STATES / 128, 256, 0, stream>>>(q_f16, kall_f16, belief, m_buf, z_buf, prior);

  k_keys<<<N_STATES / 256, 256, 0, stream>>>(prior, keys);
  for (int k = 2; k <= N_STATES; k <<= 1)
    for (int j = k >> 1; j > 0; j >>= 1)
      k_bitonic<<<N_STATES / 256, 256, 0, stream>>>(keys, j, k);

  k_sparsemax<<<1, 1024, 0, stream>>>(keys, out_vals, out_bits);
}
